// AttentionRNNScratch_69655779606699
// MI455X (gfx1250) — matrix operand order repaired and checked by reading, not compiled, not run on hardware
//
#include <hip/hip_runtime.h>
#include <hip/hip_bf16.h>

#define VOCAB 64
#define HID   128
#define BB    256
#define TT    512
#define TPAD  544   // TT + 32: lets diagonal attention chunks read past T in-bounds

typedef __attribute__((ext_vector_type(16))) _Float16 v16h;
typedef __attribute__((ext_vector_type(8)))  float    v8f;

__device__ __forceinline__ v8f wmma_f16(v16h a, v16h b, v8f c) {
  return __builtin_amdgcn_wmma_f32_16x16x32_f16(false, a, false, b, (short)0, c, false, false);
}

__device__ __forceinline__ float fast_exp(float x) {
  return __builtin_amdgcn_exp2f(x * 1.4426950408889634f);
}
__device__ __forceinline__ float fast_tanh(float x) {
  // tanh(x) = 1 - 2/(e^{2x}+1); exp2(+inf)->inf -> 1 ; exp2(-inf)->0 -> -1
  float e = __builtin_amdgcn_exp2f(x * 2.8853900817779268f);
  return 1.0f - 2.0f * __builtin_amdgcn_rcpf(e + 1.0f);
}

// ---------------- prep: fp32 -> fp16 weight conversion ----------------
__global__ void prep_kernel(const float* emb, const float* Wx, const float* Wh,
                            const float* Wout, _Float16* emb_h, _Float16* wx_h,
                            _Float16* wh_h, _Float16* wout_h) {
  int i = blockIdx.x * 256 + threadIdx.x;
  if (i < VOCAB * HID) emb_h[i] = (_Float16)emb[i];
  if (i < HID * HID) { wx_h[i] = (_Float16)Wx[i]; wh_h[i] = (_Float16)Wh[i]; }
  if (i < VOCAB * 2 * HID) wout_h[i] = (_Float16)Wout[i];
}

// zero the [TT, TPAD) tail of every hbufT row so overhang reads are finite
__global__ void pad_kernel(_Float16* hbufT) {
  int i = blockIdx.x * 256 + threadIdx.x;       // over BB*HID*(TPAD-TT)
  int row = i >> 5, col = TT + (i & 31);
  hbufT[(size_t)row * TPAD + col] = (_Float16)0.0f;
}

// ---------------- pass 1: xproj[t][b][j] = emb[x[b,t]] @ Wx.T + bx ----------------
__global__ void xproj_kernel(const int* __restrict__ x, const float* __restrict__ bx,
                             const _Float16* __restrict__ emb_h,
                             const _Float16* __restrict__ wx_h,
                             float* __restrict__ xproj) {
  int lane = threadIdx.x & 31, w = threadIdx.x >> 5;
  int hl = lane >> 4, ln = lane & 15;
  int mt = blockIdx.x;
  int r = mt * 16 + ln;          // A-matrix row owned by this lane
  int idx = x[r];
  int j = w * 16 + ln;           // output column (B-matrix N)
  v8f c = {};
#pragma unroll
  for (int kt = 0; kt < 4; kt++) {
    v16h a, bf;
    const _Float16* ap = emb_h + idx * HID + kt * 32;
#pragma unroll
    for (int e = 0; e < 8; e++) { a[e] = ap[16 * hl + e]; a[8 + e] = ap[16 * hl + 8 + e]; }
    const _Float16* bp = wx_h + j * HID + kt * 32 + 16 * hl;
#pragma unroll
    for (int e = 0; e < 16; e++) bf[e] = bp[e];
    c = wmma_f16(a, bf, c);
  }
  float bj = bx[j];
#pragma unroll
  for (int v = 0; v < 8; v++) {
    int rr = mt * 16 + v + 8 * hl;        // C row M = v + 8*hl
    int b = rr >> 9, t = rr & (TT - 1);   // T = 512, tiles never cross b
    xproj[((size_t)t * BB + b) * HID + j] = c[v] + bj;
  }
}

// ---------------- pass 2: serial scan  h = tanh(xproj_t + h @ Wh.T) ----------------
// grid: 16 blocks (16 batches each) x 256 threads (8 waves = 8 N-tiles)
__global__ void scan_kernel(const float* __restrict__ h0,
                            const _Float16* __restrict__ wh_h,
                            const float* __restrict__ xproj,
                            _Float16* __restrict__ hbuf, _Float16* __restrict__ hbufT,
                            float* __restrict__ out_h) {
  __shared__ _Float16 hs[2][16][136];
  int lane = threadIdx.x & 31, w = threadIdx.x >> 5;
  int hl = lane >> 4, ln = lane & 15;
  int b0 = blockIdx.x * 16;
  int j = w * 16 + ln;
  // Wh B-fragments are loop-invariant: keep all 4 K-steps in VGPRs (32 VGPRs)
  v16h whf[4];
#pragma unroll
  for (int kt = 0; kt < 4; kt++) {
    const _Float16* bp = wh_h + j * HID + kt * 32 + 16 * hl;
#pragma unroll
    for (int e = 0; e < 16; e++) whf[kt][e] = bp[e];
  }
  for (int i = threadIdx.x; i < 16 * HID; i += 256) {
    int m = i >> 7, k = i & (HID - 1);
    hs[0][m][k] = (_Float16)h0[(size_t)(b0 + m) * HID + k];
  }
  // software pipeline: xp addend for step t loaded during step t-1
  float xp[8];
#pragma unroll
  for (int v = 0; v < 8; v++)
    xp[v] = xproj[((size_t)0 * BB + b0 + v + 8 * hl) * HID + j];
  __syncthreads();
  int cur = 0;
  for (int t = 0; t < TT; t++) {
    // issue next step's loads immediately; latency spans WMMA chain + barrier
    float xpn[8];
    int tn = (t < TT - 1) ? t + 1 : t;
#pragma unroll
    for (int v = 0; v < 8; v++)
      xpn[v] = xproj[((size_t)tn * BB + b0 + v + 8 * hl) * HID + j];
    // two independent accumulator chains to overlap XDL latency
    v8f c0 = {}, c1 = {};
#pragma unroll
    for (int kt = 0; kt < 4; kt++) {
      v16h a;
#pragma unroll
      for (int e = 0; e < 8; e++) {
        a[e]     = hs[cur][ln][kt * 32 + 16 * hl + e];
        a[8 + e] = hs[cur][ln][kt * 32 + 16 * hl + 8 + e];
      }
      if (kt & 1) c1 = wmma_f16(a, whf[kt], c1);
      else        c0 = wmma_f16(a, whf[kt], c0);
    }
    int nxt = cur ^ 1;
#pragma unroll
    for (int v = 0; v < 8; v++) {
      int m = v + 8 * hl;
      float hv = fast_tanh(c0[v] + c1[v] + xp[v]);
      _Float16 hh = (_Float16)hv;
      hs[nxt][m][j] = hh;
      size_t b = b0 + m;
      hbuf[(b * TT + t) * HID + j] = hh;
      hbufT[(b * HID + j) * TPAD + t] = hh;
      if (t == TT - 1) out_h[b * HID + j] = hv;
      xp[v] = xpn[v];
    }
    __syncthreads();
    cur = nxt;
  }
}

// ---------------- pass 3: causal flash attention over hbuf (Q=K=V=h) ----------------
// grid: (T/128, B) blocks x 256 threads; wave w owns Q-tile t0=(bx*8+w)*16
__global__ void attn_kernel(const _Float16* __restrict__ hbuf,
                            const _Float16* __restrict__ hbufT,
                            _Float16* __restrict__ ctxb) {
  __shared__ _Float16 ps[8][16][40];
  int lane = threadIdx.x & 31, w = threadIdx.x >> 5;
  int hl = lane >> 4, ln = lane & 15;
  int b = blockIdx.y;
  int t0 = (blockIdx.x * 8 + w) * 16;
  // Q fragments resident (32 VGPRs)
  v16h qf[4];
  const _Float16* qbase = hbuf + ((size_t)b * TT + t0 + ln) * HID;
#pragma unroll
  for (int kt = 0; kt < 4; kt++) {
#pragma unroll
    for (int e = 0; e < 8; e++) {
      qf[kt][e]     = qbase[kt * 32 + 16 * hl + e];
      qf[kt][8 + e] = qbase[kt * 32 + 16 * hl + 8 + e];
    }
  }
  v8f ctx[8];
#pragma unroll
  for (int nt = 0; nt < 8; nt++) ctx[nt] = (v8f){};
  float m[8], l[8];
#pragma unroll
  for (int v = 0; v < 8; v++) { m[v] = -1e30f; l[v] = 0.0f; }

  for (int cs = 0; cs <= t0 + 15; cs += 32) {
    // S = Q K^T : two 16x16 C tiles covering 32 s-columns
    v8f S[2];
#pragma unroll
    for (int n = 0; n < 2; n++) {
      v8f c = {};
      int s = cs + 16 * n + ln; if (s > TT - 1) s = TT - 1;   // row-pointer clamp only
      const _Float16* kp = hbuf + ((size_t)b * TT + s) * HID;
#pragma unroll
      for (int kt = 0; kt < 4; kt++) {
        v16h bf;
#pragma unroll
        for (int e = 0; e < 16; e++) bf[e] = kp[kt * 32 + 16 * hl + e];
        c = wmma_f16(qf[kt], bf, c);
      }
      S[n] = c;
    }
    // causal mask + running softmax (row M = v + 8*hl lives in one 16-lane half)
    float sc[8];
#pragma unroll
    for (int v = 0; v < 8; v++) {
      int trow = t0 + v + 8 * hl;
#pragma unroll
      for (int n = 0; n < 2; n++) {
        int s = cs + 16 * n + ln;
        if (s > trow) S[n][v] = -1e30f;
      }
      float a = fmaxf(S[0][v], S[1][v]);
#pragma unroll
      for (int off = 1; off < 16; off <<= 1) a = fmaxf(a, __shfl_xor(a, off, 32));
      float mn = fmaxf(m[v], a);
      sc[v] = fast_exp(m[v] - mn);
      m[v] = mn;
      float p0 = fast_exp(S[0][v] - mn);
      float p1 = fast_exp(S[1][v] - mn);
      S[0][v] = p0; S[1][v] = p1;
      float r = p0 + p1;
#pragma unroll
      for (int off = 1; off < 16; off <<= 1) r += __shfl_xor(r, off, 32);
      l[v] = l[v] * sc[v] + r;
    }
#pragma unroll
    for (int nt = 0; nt < 8; nt++)
#pragma unroll
      for (int v = 0; v < 8; v++) ctx[nt][v] *= sc[v];
    // P: C-layout -> A-layout via LDS bounce (same-wave DS ops are in-order)
#pragma unroll
    for (int v = 0; v < 8; v++) {
      int mr = v + 8 * hl;
      ps[w][mr][ln]      = (_Float16)S[0][v];
      ps[w][mr][16 + ln] = (_Float16)S[1][v];
    }
    asm volatile("s_wait_dscnt 0" ::: "memory");
    v16h pf;
#pragma unroll
    for (int e = 0; e < 8; e++) {
      pf[e]     = ps[w][ln][16 * hl + e];
      pf[8 + e] = ps[w][ln][16 * hl + 8 + e];
    }
    // ctx += P V : V fragments are contiguous b128 loads (TPAD absorbs overhang)
#pragma unroll
    for (int nt = 0; nt < 8; nt++) {
      v16h vf;
      int hcol = nt * 16 + ln;
      const _Float16* vp = hbufT + ((size_t)b * HID + hcol) * TPAD + cs + 16 * hl;
#pragma unroll
      for (int e = 0; e < 16; e++) vf[e] = vp[e];
      ctx[nt] = wmma_f16(pf, vf, ctx[nt]);
    }
  }
#pragma unroll
  for (int v = 0; v < 8; v++) l[v] = __builtin_amdgcn_rcpf(l[v]);
#pragma unroll
  for (int nt = 0; nt < 8; nt++)
#pragma unroll
    for (int v = 0; v < 8; v++)
      ctxb[((size_t)b * TT + t0 + v + 8 * hl) * HID + nt * 16 + ln] =
          (_Float16)(ctx[nt][v] * l[v]);
}

// ---------------- pass 4: logits = [h | ctx] @ Wout.T + bout ----------------
// grid: (B*T)/16 blocks of 128 threads (4 waves = 4 vocab N-tiles)
__global__ void logits_kernel(const _Float16* __restrict__ hbuf,
                              const _Float16* __restrict__ ctxb,
                              const _Float16* __restrict__ wout_h,
                              const float* __restrict__ bout,
                              float* __restrict__ out) {
  int lane = threadIdx.x & 31, w = threadIdx.x >> 5;
  int hl = lane >> 4, ln = lane & 15;
  int mt = blockIdx.x;
  int r = mt * 16 + ln;
  int vcol = w * 16 + ln;
  v8f c0 = {}, c1 = {};   // two chains: h-half and ctx-half overlap
#pragma unroll
  for (int kt = 0; kt < 8; kt++) {
    const _Float16* asrc = (kt < 4) ? (hbuf + (size_t)r * HID + kt * 32)
                                    : (ctxb + (size_t)r * HID + (kt - 4) * 32);
    v16h a, bf;
#pragma unroll
    for (int e = 0; e < 8; e++) {
      a[e]     = asrc[16 * hl + e];
      a[8 + e] = asrc[16 * hl + 8 + e];
    }
    const _Float16* bp = wout_h + vcol * (2 * HID) + kt * 32 + 16 * hl;
#pragma unroll
    for (int e = 0; e < 16; e++) bf[e] = bp[e];
    if (kt < 4) c0 = wmma_f16(a, bf, c0);
    else        c1 = wmma_f16(a, bf, c1);
  }
  float bo = bout[vcol];
#pragma unroll
  for (int v = 0; v < 8; v++)
    out[(size_t)(mt * 16 + v + 8 * hl) * VOCAB + vcol] = c0[v] + c1[v] + bo;
}

extern "C" void kernel_launch(void* const* d_in, const int* in_sizes, int n_in,
                              void* d_out, int out_size, void* d_ws, size_t ws_size,
                              hipStream_t stream) {
  const int*   x    = (const int*)  d_in[0];
  const float* h0   = (const float*)d_in[1];
  const float* emb  = (const float*)d_in[2];
  const float* Wx   = (const float*)d_in[3];
  const float* bx   = (const float*)d_in[4];
  const float* Wh   = (const float*)d_in[5];
  const float* Wout = (const float*)d_in[6];
  const float* bout = (const float*)d_in[7];

  char* ws = (char*)d_ws;
  _Float16* emb_h  = (_Float16*)ws;                    // 64*128
  _Float16* wx_h   = emb_h + VOCAB * HID;              // 128*128
  _Float16* wh_h   = wx_h + HID * HID;                 // 128*128
  _Float16* wout_h = wh_h + HID * HID;                 // 64*256
  float*    xproj  = (float*)(ws + (1 << 17));                 // [T][B][H] f32, 64 MB
  _Float16* hbuf   = (_Float16*)(ws + (1 << 17) +
                                 (size_t)TT * BB * HID * 4);   // [B][T][H] f16
  _Float16* hbufT  = hbuf + (size_t)BB * TT * HID;             // [B][H][TPAD] f16
  _Float16* ctxb   = hbufT + (size_t)BB * HID * TPAD;          // [B][T][H] f16

  float* out_logits = (float*)d_out;
  float* out_h = out_logits + (size_t)BB * TT * VOCAB;

  prep_kernel<<<64, 256, 0, stream>>>(emb, Wx, Wh, Wout, emb_h, wx_h, wh_h, wout_h);
  pad_kernel<<<(BB * HID * (TPAD - TT)) / 256, 256, 0, stream>>>(hbufT);
  xproj_kernel<<<(BB * TT) / 16, 256, 0, stream>>>(x, bx, emb_h, wx_h, xproj);
  scan_kernel<<<BB / 16, 256, 0, stream>>>(h0, wh_h, xproj, hbuf, hbufT, out_h);
  attn_kernel<<<dim3(TT / 128, BB), 256, 0, stream>>>(hbuf, hbufT, ctxb);
  logits_kernel<<<(BB * TT) / 16, 128, 0, stream>>>(hbuf, ctxb, wout_h, bout, out_logits);
}
